// Net_52948356825721
// MI455X (gfx1250) — compile-verified, matrix operand order checked by reading, not run on hardware
//
#include <hip/hip_runtime.h>

typedef _Float16 v16h __attribute__((ext_vector_type(16)));
typedef _Float16 v8h  __attribute__((ext_vector_type(8)));
typedef float    v8f  __attribute__((ext_vector_type(8)));

// ---------------------------------------------------------------------------
// compile-time for-loop
// ---------------------------------------------------------------------------
template <int V> struct Ic { static constexpr int value = V; };

template <int N, int I = 0, typename F>
__device__ __forceinline__ void static_for(F&& f) {
    if constexpr (I < N) {
        f(Ic<I>{});
        static_for<N, I + 1>(static_cast<F&&>(f));
    }
}

// ---------------------------------------------------------------------------
// Static per-conv layout of the concatenated per-edge A matrix.
// sel: 0 = gather with row[e], 1 = gather with col[e], 2 = per-edge array.
// All region starts are multiples of 4 -> a 4-aligned quad never straddles.
// ---------------------------------------------------------------------------
template <int CONV> struct ConvCfg;
template <> struct ConvCfg<0> {          // conv1: [nf[row] | nf[col] | edge_attr | edge_rot]
    static constexpr int nreg = 6, ktrue = 1164, nchunks = 37;
    static constexpr int start[7]  = {0, 256, 260, 516, 520, 1160, 1164};
    static constexpr int stride[6] = {256, 4, 256, 4, 640, 4};
    static constexpr int sel[6]    = {0, 0, 1, 1, 2, 2};
};
template <> struct ConvCfg<1> {          // conv2: [x1[row] | x1[col] | edge_attr | e1]
    static constexpr int nreg = 4, ktrue = 832, nchunks = 26;
    static constexpr int start[7]  = {0, 64, 128, 768, 832, 832, 832};
    static constexpr int stride[6] = {64, 64, 640, 64, 1, 1};
    static constexpr int sel[6]    = {0, 1, 2, 2, 2, 2};
};
template <> struct ConvCfg<2> {          // conv3/conv4: [a[row]|b[row]|a[col]|b[col]|ea|eb]
    static constexpr int nreg = 6, ktrue = 384, nchunks = 12;
    static constexpr int start[7]  = {0, 64, 128, 192, 256, 320, 384};
    static constexpr int stride[6] = {64, 64, 64, 64, 64, 64};
    static constexpr int sel[6]    = {0, 0, 1, 1, 2, 2};
};

template <typename C>
__host__ __device__ constexpr int region_of(int q) {
    for (int i = 0; i < C::nreg; ++i)
        if (q >= C::start[i] && q < C::start[i + 1]) return i;
    return 0;
}

// Quad loader with compile-time region resolution for both lane halves.
// Q0 = quad for lanes 0-15; Q0+8 = quad for lanes 16-31 (hi = lane>>4).
template <int CONV, int Q0>
__device__ __forceinline__ float4 load_quad2(const float* const bases[6],
                                             int ri, int ci, int e, int hi) {
    using C = ConvCfg<CONV>;
    float4 z = make_float4(0.f, 0.f, 0.f, 0.f);
    if constexpr (Q0 >= C::ktrue) {
        return z;                                     // both halves are zero padding
    } else {
        constexpr int r0 = region_of<C>(Q0);
        const int idx0 = (C::sel[r0] == 0) ? ri : ((C::sel[r0] == 1) ? ci : e);
        const float* p0 = bases[r0] + (size_t)idx0 * C::stride[r0] + (Q0 - C::start[r0]);
        if constexpr (Q0 + 8 >= C::ktrue) {
            float4 v = z;                             // only low half is in range
            if (hi == 0) v = *(const float4*)p0;
            return v;
        } else {
            constexpr int r1 = region_of<C>(Q0 + 8);
            if constexpr (r0 == r1) {
                return *(const float4*)(p0 + (hi << 3));   // common case: one load, imm offset
            } else {
                const int idx1 = (C::sel[r1] == 0) ? ri : ((C::sel[r1] == 1) ? ci : e);
                const float* p1 = bases[r1] + (size_t)idx1 * C::stride[r1]
                                + (Q0 + 8 - C::start[r1]);
                const float* p = hi ? p1 : p0;
                return *(const float4*)p;
            }
        }
    }
}

// ---------------------------------------------------------------------------
// Weight pre-swizzle: W[K x 64] f32 -> f16 B-fragments in WMMA lane order.
// k = c*32 + (lane>>4)*8 + (j<8 ? j : 8+j),  n = t*16 + (lane&15)
// ---------------------------------------------------------------------------
__global__ void swizzle_w_kernel(const float* __restrict__ W, _Float16* __restrict__ out,
                                 int Ktrue, int nchunks) {
    int idx = blockIdx.x * blockDim.x + threadIdx.x;
    int total = nchunks * 4 * 32 * 16;
    if (idx >= total) return;
    int j    = idx & 15;
    int lane = (idx >> 4) & 31;
    int ct   = idx >> 9;
    int c    = ct >> 2;
    int t    = ct & 3;
    int k    = c * 32 + (lane >> 4) * 8 + (j < 8 ? j : 8 + j);
    int n    = t * 16 + (lane & 15);
    float v  = (k < Ktrue) ? W[(size_t)k * 64 + n] : 0.f;
    out[idx] = (_Float16)v;
}

// ---------------------------------------------------------------------------
// Edge-conv: one wave handles MT consecutive 16-edge A-tiles (B reuse = MT).
// Layer 1: H = relu(A[16xK] @ W1 + b1) via fully unrolled K/32 x 4 WMMAs
// Layer 2: M = H @ W2 + b2 via 2 x 4 WMMAs (A re-laid-out through LDS)
// Then: edge_out = (relu?)M ; node_acc[row[e]] += M (f32 atomics)
// ---------------------------------------------------------------------------
template <int CONV, int MT>
__global__ __launch_bounds__(256) void edge_conv_kernel(
    const float* b0, const float* b1r, const float* b2r, const float* b3r,
    const float* b4r, const float* b5r,
    const int* __restrict__ row, const int* __restrict__ col,
    const _Float16* __restrict__ w1f, const float* __restrict__ bias1,
    const _Float16* __restrict__ w2f, const float* __restrict__ bias2,
    float* __restrict__ edge_out, float* __restrict__ node_acc,
    int relu_edge, int nE)
{
    using C = ConvCfg<CONV>;
    __shared__ __align__(16) _Float16 lds_h[8][MT][16 * 64];

    const int lane  = threadIdx.x & 31;
    const int wv    = threadIdx.x >> 5;
    const int ebase = (blockIdx.x * 8 + wv) * (16 * MT);
    if (ebase >= nE) return;

    const int ln = lane & 15;
    const int hi = lane >> 4;

    const float* bases[6] = {b0, b1r, b2r, b3r, b4r, b5r};

    int eT[MT], riT[MT], ciT[MT];
#pragma unroll
    for (int T = 0; T < MT; ++T) {
        eT[T]  = ebase + T * 16 + ln;
        riT[T] = row[eT[T]];
        ciT[T] = col[eT[T]];
    }

    v8f acc[MT][4];
#pragma unroll
    for (int T = 0; T < MT; ++T)
#pragma unroll
        for (int t = 0; t < 4; ++t)
#pragma unroll
            for (int j = 0; j < 8; ++j) acc[T][t][j] = 0.f;

    // ---- layer 1: fully unrolled over K chunks; region search folds away ----
    static_for<C::nchunks>([&](auto cI) {
        constexpr int c = decltype(cI)::value;
        v16h a[MT];
        static_for<MT>([&](auto tI) {
            constexpr int T = decltype(tI)::value;
            float4 q0 = load_quad2<CONV, c * 32 + 0 >(bases, riT[T], ciT[T], eT[T], hi);
            float4 q1 = load_quad2<CONV, c * 32 + 4 >(bases, riT[T], ciT[T], eT[T], hi);
            float4 q2 = load_quad2<CONV, c * 32 + 16>(bases, riT[T], ciT[T], eT[T], hi);
            float4 q3 = load_quad2<CONV, c * 32 + 20>(bases, riT[T], ciT[T], eT[T], hi);
            a[T][0]  = (_Float16)q0.x; a[T][1]  = (_Float16)q0.y;
            a[T][2]  = (_Float16)q0.z; a[T][3]  = (_Float16)q0.w;
            a[T][4]  = (_Float16)q1.x; a[T][5]  = (_Float16)q1.y;
            a[T][6]  = (_Float16)q1.z; a[T][7]  = (_Float16)q1.w;
            a[T][8]  = (_Float16)q2.x; a[T][9]  = (_Float16)q2.y;
            a[T][10] = (_Float16)q2.z; a[T][11] = (_Float16)q2.w;
            a[T][12] = (_Float16)q3.x; a[T][13] = (_Float16)q3.y;
            a[T][14] = (_Float16)q3.z; a[T][15] = (_Float16)q3.w;
        });
        static_for<4>([&](auto nI) {
            constexpr int nt = decltype(nI)::value;
            const v16h b = *(const v16h*)(w1f + ((((size_t)c * 4 + nt) * 32 + lane) << 4));
            static_for<MT>([&](auto tI) {
                constexpr int T = decltype(tI)::value;
                acc[T][nt] = __builtin_amdgcn_wmma_f32_16x16x32_f16(
                    false, a[T], false, b, (short)0, acc[T][nt], false, false);
            });
        });
    });

    // ---- bias + relu, stage H into LDS (C-layout -> row-major f16) ----
    static_for<4>([&](auto nI) {
        constexpr int nt = decltype(nI)::value;
        const int cc   = nt * 16 + ln;
        const float bv = bias1[cc];
        static_for<MT>([&](auto tI) {
            constexpr int T = decltype(tI)::value;
#pragma unroll
            for (int v = 0; v < 8; ++v) {
                const int m = v + 8 * hi;
                float h = acc[T][nt][v] + bv;
                lds_h[wv][T][m * 64 + cc] = (_Float16)fmaxf(h, 0.f);
            }
        });
    });
    __syncthreads();

    // ---- layer 2: K = 64 (2 chunks) ----
    v8f acc2[MT][4];
#pragma unroll
    for (int T = 0; T < MT; ++T)
#pragma unroll
        for (int t = 0; t < 4; ++t)
#pragma unroll
            for (int j = 0; j < 8; ++j) acc2[T][t][j] = 0.f;

    static_for<2>([&](auto cI) {
        constexpr int c2 = decltype(cI)::value;
        const int kb = c2 * 32 + hi * 8;
        v16h a2[MT];
        static_for<MT>([&](auto tI) {
            constexpr int T = decltype(tI)::value;
            v8h lo = *(const v8h*)&lds_h[wv][T][ln * 64 + kb];
            v8h hh = *(const v8h*)&lds_h[wv][T][ln * 64 + kb + 16];
#pragma unroll
            for (int j = 0; j < 8; ++j) { a2[T][j] = lo[j]; a2[T][8 + j] = hh[j]; }
        });
        static_for<4>([&](auto nI) {
            constexpr int nt = decltype(nI)::value;
            const v16h b = *(const v16h*)(w2f + ((((size_t)c2 * 4 + nt) * 32 + lane) << 4));
            static_for<MT>([&](auto tI) {
                constexpr int T = decltype(tI)::value;
                acc2[T][nt] = __builtin_amdgcn_wmma_f32_16x16x32_f16(
                    false, a2[T], false, b, (short)0, acc2[T][nt], false, false);
            });
        });
    });

    // ---- epilogue: bias, store edge features, scatter-sum onto nodes ----
    static_for<MT>([&](auto tI) {
        constexpr int T = decltype(tI)::value;
        const int eb = ebase + T * 16;
        int rI[8];
#pragma unroll
        for (int v = 0; v < 8; ++v) rI[v] = row[eb + v + 8 * hi];
        static_for<4>([&](auto nI) {
            constexpr int nt = decltype(nI)::value;
            const int cc   = nt * 16 + ln;
            const float bv = bias2[cc];
#pragma unroll
            for (int v = 0; v < 8; ++v) {
                const int m  = v + 8 * hi;
                const float o = acc2[T][nt][v] + bv;
                edge_out[(size_t)(eb + m) * 64 + cc] = relu_edge ? fmaxf(o, 0.f) : o;
                atomicAdd(&node_acc[(size_t)rI[v] * 64 + cc], o);
            }
        });
    });
}

// ---------------------------------------------------------------------------
// Helpers
// ---------------------------------------------------------------------------
__global__ void zero_f32_kernel(float* p, int n) {
    int i = blockIdx.x * blockDim.x + threadIdx.x;
    if (i < n) p[i] = 0.f;
}
__global__ void zero_i32_kernel(int* p, int n) {
    int i = blockIdx.x * blockDim.x + threadIdx.x;
    if (i < n) p[i] = 0;
}
__global__ void deg_kernel(const int* __restrict__ row, int* __restrict__ cnt, int nE) {
    int i = blockIdx.x * blockDim.x + threadIdx.x;
    if (i < nE) atomicAdd(&cnt[row[i]], 1);
}
__global__ void node_finalize_kernel(const int* __restrict__ cnt, float* __restrict__ acc,
                                     float* __restrict__ xout, int do_relu, int total) {
    int i = blockIdx.x * blockDim.x + threadIdx.x;
    if (i >= total) return;
    float cv = (float)cnt[i >> 6];
    float v  = acc[i] / fmaxf(cv, 1.f);
    if (do_relu) v = fmaxf(v, 0.f);
    xout[i] = v;
    acc[i]  = 0.f;
}
__global__ void lin1_norm_kernel(const float* __restrict__ x4, const float* __restrict__ w,
                                 const float* __restrict__ b, float* __restrict__ out, int N) {
    int n = blockIdx.x * blockDim.x + threadIdx.x;
    if (n >= N) return;
    float o0 = b[0], o1 = b[1], o2 = b[2], o3 = b[3];
#pragma unroll 8
    for (int k = 0; k < 64; ++k) {
        float xv = x4[(size_t)n * 64 + k];
        o0 += xv * w[k * 4 + 0];
        o1 += xv * w[k * 4 + 1];
        o2 += xv * w[k * 4 + 2];
        o3 += xv * w[k * 4 + 3];
    }
    float nm = fmaxf(sqrtf(o0 * o0 + o1 * o1 + o2 * o2 + o3 * o3), 1e-12f);
    float inv = 1.f / nm;
    out[(size_t)n * 4 + 0] = o0 * inv;
    out[(size_t)n * 4 + 1] = o1 * inv;
    out[(size_t)n * 4 + 2] = o2 * inv;
    out[(size_t)n * 4 + 3] = o3 * inv;
}
__global__ void edge_fc_kernel(const float* __restrict__ e4, const float* __restrict__ w1,
                               const float* __restrict__ b1, const float* __restrict__ w2,
                               const float* __restrict__ b2, float* __restrict__ out, int H) {
    int i = blockIdx.x * blockDim.x + threadIdx.x;
    if (i >= H) return;
    float h[64];
#pragma unroll 8
    for (int k = 0; k < 64; ++k)
        h[k] = e4[(size_t)i * 64 + k] + e4[(size_t)(i + H) * 64 + k];
    float o = b2[0];
    for (int j = 0; j < 32; ++j) {
        float a = b1[j];
#pragma unroll 8
        for (int k = 0; k < 64; ++k) a += h[k] * w1[k * 32 + j];
        o += fmaxf(a, 0.f) * w2[j];
    }
    out[i] = o;
}

// ---------------------------------------------------------------------------
// Host orchestration
// ---------------------------------------------------------------------------
extern "C" void kernel_launch(void* const* d_in, const int* in_sizes, int n_in,
                              void* d_out, int out_size, void* d_ws, size_t ws_size,
                              hipStream_t stream) {
    const float* x_org     = (const float*)d_in[0];
    const float* x_rot     = (const float*)d_in[1];
    const int*   eidx      = (const int*)d_in[2];
    const float* edge_attr = (const float*)d_in[3];
    const float* edge_rot  = (const float*)d_in[4];
    const float* c_w1[4] = {(const float*)d_in[5],  (const float*)d_in[9],
                            (const float*)d_in[13], (const float*)d_in[17]};
    const float* c_b1[4] = {(const float*)d_in[6],  (const float*)d_in[10],
                            (const float*)d_in[14], (const float*)d_in[18]};
    const float* c_w2[4] = {(const float*)d_in[7],  (const float*)d_in[11],
                            (const float*)d_in[15], (const float*)d_in[19]};
    const float* c_b2[4] = {(const float*)d_in[8],  (const float*)d_in[12],
                            (const float*)d_in[16], (const float*)d_in[20]};
    const float* lin1_w = (const float*)d_in[21];
    const float* lin1_b = (const float*)d_in[22];
    const float* efc_w1 = (const float*)d_in[23];
    const float* efc_b1 = (const float*)d_in[24];
    const float* efc_w2 = (const float*)d_in[25];
    const float* efc_b2 = (const float*)d_in[26];

    const int N = in_sizes[0] / 256;   // 10000
    const int E = in_sizes[2] / 2;     // 128000
    const int H = E / 2;
    const int* rowp = eidx;
    const int* colp = eidx + E;
    float* dout = (float*)d_out;

    // ---- carve workspace ----
    char* wp = (char*)d_ws;
    auto carve = [&](size_t bytes) -> void* {
        void* r = (void*)wp;
        wp += (bytes + 255) & ~(size_t)255;
        return r;
    };
    int*   cnt  = (int*)  carve((size_t)N * sizeof(int));
    float* accN = (float*)carve((size_t)N * 64 * sizeof(float));
    float* x1   = (float*)carve((size_t)N * 64 * sizeof(float));
    float* x2   = (float*)carve((size_t)N * 64 * sizeof(float));
    float* x3   = (float*)carve((size_t)N * 64 * sizeof(float));
    float* x4   = (float*)carve((size_t)N * 64 * sizeof(float));
    float* e1   = (float*)carve((size_t)E * 64 * sizeof(float));
    float* e2   = (float*)carve((size_t)E * 64 * sizeof(float));
    float* e3   = (float*)carve((size_t)E * 64 * sizeof(float));
    float* e4   = (float*)carve((size_t)E * 64 * sizeof(float));

    const int Ktrue1[4]  = {1164, 832, 384, 384};
    const int nchunk1[4] = {37, 26, 12, 12};
    _Float16* w1f[4];
    _Float16* w2f[4];
    for (int i = 0; i < 4; ++i) {
        w1f[i] = (_Float16*)carve((size_t)nchunk1[i] * 4 * 32 * 16 * sizeof(_Float16));
        w2f[i] = (_Float16*)carve((size_t)2 * 4 * 32 * 16 * sizeof(_Float16));
    }

    zero_i32_kernel<<<(N + 255) / 256, 256, 0, stream>>>(cnt, N);
    zero_f32_kernel<<<(N * 64 + 255) / 256, 256, 0, stream>>>(accN, N * 64);
    deg_kernel<<<(E + 255) / 256, 256, 0, stream>>>(rowp, cnt, E);

    for (int i = 0; i < 4; ++i) {
        int t1 = nchunk1[i] * 4 * 32 * 16;
        swizzle_w_kernel<<<(t1 + 255) / 256, 256, 0, stream>>>(c_w1[i], w1f[i], Ktrue1[i], nchunk1[i]);
        int t2 = 2 * 4 * 32 * 16;
        swizzle_w_kernel<<<(t2 + 255) / 256, 256, 0, stream>>>(c_w2[i], w2f[i], 64, 2);
    }

    constexpr int MT = 2;                      // 16-edge tiles per wave (B reuse)
    const int tiles16 = (E + 15) / 16;
    const int wtiles  = (tiles16 + MT - 1) / MT;
    const int blocks  = (wtiles + 7) / 8;
    const int fin_b   = (N * 64 + 255) / 256;

    // conv1
    edge_conv_kernel<0, MT><<<blocks, 256, 0, stream>>>(
        x_org, x_rot, x_org, x_rot, edge_attr, edge_rot,
        rowp, colp, w1f[0], c_b1[0], w2f[0], c_b2[0], e1, accN, 1, E);
    node_finalize_kernel<<<fin_b, 256, 0, stream>>>(cnt, accN, x1, 0, N * 64);
    // conv2
    edge_conv_kernel<1, MT><<<blocks, 256, 0, stream>>>(
        x1, x1, edge_attr, e1, e1, e1,
        rowp, colp, w1f[1], c_b1[1], w2f[1], c_b2[1], e2, accN, 1, E);
    node_finalize_kernel<<<fin_b, 256, 0, stream>>>(cnt, accN, x2, 1, N * 64);
    // conv3
    edge_conv_kernel<2, MT><<<blocks, 256, 0, stream>>>(
        x2, x1, x2, x1, e2, e1,
        rowp, colp, w1f[2], c_b1[2], w2f[2], c_b2[2], e3, accN, 1, E);
    node_finalize_kernel<<<fin_b, 256, 0, stream>>>(cnt, accN, x3, 1, N * 64);
    // conv4
    edge_conv_kernel<2, MT><<<blocks, 256, 0, stream>>>(
        x3, x2, x3, x2, e3, e2,
        rowp, colp, w1f[3], c_b1[3], w2f[3], c_b2[3], e4, accN, 0, E);
    node_finalize_kernel<<<fin_b, 256, 0, stream>>>(cnt, accN, x4, 1, N * 64);

    // heads
    lin1_norm_kernel<<<(N + 255) / 256, 256, 0, stream>>>(x4, lin1_w, lin1_b, dout, N);
    edge_fc_kernel<<<(H + 255) / 256, 256, 0, stream>>>(e4, efc_w1, efc_b1, efc_w2, efc_b2,
                                                        dout + (size_t)N * 4, H);
}
